// HelixLinear_47201690583539
// MI455X (gfx1250) — matrix operand order repaired and checked by reading, not compiled, not run on hardware
//
#include <hip/hip_runtime.h>
#include <hip/hip_bf16.h>

// ---------------------------------------------------------------------------
// HelixLinear fused VQ-dequant GEMM for MI455X (gfx1250, wave32, WMMA).
//   y[m,o] = sum_i x[m,i] * (codebook[idx[o,i]]) + bias[o]
//            + lowrank (rank 32) + sidecar scatter
// Roofline: indices = 256 MB (dominant) -> ~11us floor @ 23.3 TB/s.
// Strategy: read indices exactly once, dequant -> bf16 B fragments in LDS,
// v_wmma_f32_16x16x32_bf16, x kept bf16 in L2.
// ---------------------------------------------------------------------------

typedef __attribute__((ext_vector_type(16))) __bf16 v16bf;
typedef __attribute__((ext_vector_type(8)))  float  v8f;

struct Bits256 { uint4 lo, hi; };   // 32 bytes -> one v16bf fragment

__device__ __forceinline__ unsigned short f2bf(float f) {
    unsigned int u = __float_as_uint(f);
    u += 0x7FFFu + ((u >> 16) & 1u);          // round-to-nearest-even
    return (unsigned short)(u >> 16);
}

// ---------------- x fp32 -> bf16 (workspace, stays hot in 192MB L2) --------
__global__ void __launch_bounds__(256)
k_cvt(const float* __restrict__ x, unsigned short* __restrict__ xb) {
    const int i = (blockIdx.x * 256 + threadIdx.x) * 4;
    const float4 v = *(const float4*)(x + i);
    unsigned int d0 = (unsigned int)f2bf(v.x) | ((unsigned int)f2bf(v.y) << 16);
    unsigned int d1 = (unsigned int)f2bf(v.z) | ((unsigned int)f2bf(v.w) << 16);
    *(uint2*)(xb + i) = make_uint2(d0, d1);
}

// ---------------- T[m,r] = s[r] * sum_k x[m,k]*Vt[r,k]  (256x32, tiny) -----
__global__ void __launch_bounds__(256)
k_T(const float* __restrict__ x, const float* __restrict__ Vt,
    const float* __restrict__ s, float* __restrict__ T) {
    const int m = blockIdx.x, t = threadIdx.x;
    __shared__ float red[256];
    const float* xr = x + (size_t)m * 8192 + t * 32;
    float xv[32];
    #pragma unroll
    for (int j = 0; j < 32; j += 4) {
        float4 v = *(const float4*)(xr + j);
        xv[j] = v.x; xv[j + 1] = v.y; xv[j + 2] = v.z; xv[j + 3] = v.w;
    }
    for (int r = 0; r < 32; ++r) {
        const float* vr = Vt + (size_t)r * 8192 + t * 32;
        float p = 0.f;
        #pragma unroll
        for (int j = 0; j < 32; j += 4) {
            float4 v = *(const float4*)(vr + j);
            p = fmaf(xv[j], v.x, p); p = fmaf(xv[j + 1], v.y, p);
            p = fmaf(xv[j + 2], v.z, p); p = fmaf(xv[j + 3], v.w, p);
        }
        red[t] = p; __syncthreads();
        for (int off = 128; off > 0; off >>= 1) {
            if (t < off) red[t] += red[t + off];
            __syncthreads();
        }
        if (t == 0) T[m * 32 + r] = red[0] * s[r];
        __syncthreads();
    }
}

// ---------------- y = bias + T @ U^T  (seeds the accumulation) -------------
__global__ void __launch_bounds__(256)
k_init(const float* __restrict__ T, const float* __restrict__ U,
       const float* __restrict__ bias, float* __restrict__ y) {
    const int m = blockIdx.y;
    const int o = blockIdx.x * 256 + threadIdx.x;
    __shared__ float Tl[32];
    if (threadIdx.x < 32) Tl[threadIdx.x] = T[m * 32 + threadIdx.x];
    __syncthreads();
    float acc = bias[o];
    const float* ur = U + (size_t)o * 32;
    #pragma unroll
    for (int r = 0; r < 32; r += 4) {
        float4 u = *(const float4*)(ur + r);
        acc = fmaf(u.x, Tl[r], acc);     acc = fmaf(u.y, Tl[r + 1], acc);
        acc = fmaf(u.z, Tl[r + 2], acc); acc = fmaf(u.w, Tl[r + 3], acc);
    }
    y[(size_t)m * 8192 + o] = acc;
}

// ---------------- main fused dequant-GEMM ----------------------------------
// grid = (8192/64 N-strips, 2 K-halves), block = 256 threads = 8 waves.
// Wave w owns M rows [32w, 32w+32) x 64 cols: 2x4 accumulator tiles.
__global__ void __launch_bounds__(256)
k_gemm(const unsigned short* __restrict__ xb, const int* __restrict__ idx,
       const float* __restrict__ cbf, float* __restrict__ y) {
    __shared__ unsigned short cb[256];
    __shared__ alignas(32) unsigned short bfrag[2][4][32][16];  // 8 KB, dbl-buf

    const int t    = threadIdx.x;
    const int lane = t & 31;
    const int w    = t >> 5;
    const int n0   = blockIdx.x * 64;
    const int kbeg = blockIdx.y * 4096;
    const int kend = kbeg + 4096;

    cb[t] = f2bf(cbf[t]);                 // codebook -> bf16 in LDS
    __syncthreads();

    v8f acc[2][4];
    #pragma unroll
    for (int i = 0; i < 2; ++i)
        #pragma unroll
        for (int j = 0; j < 4; ++j)
            #pragma unroll
            for (int e = 0; e < 8; ++e) acc[i][j][e] = 0.f;

    // Dequant work item: W row n (0..63), k-octet j (0..3) of the 32-chunk.
    const int dn = t >> 2;
    const int dj = t & 3;
    const int* irow = idx + (size_t)(n0 + dn) * 8192 + dj * 8;
    // B-fragment layout: lane = n%16 + 16*((k/8)%2), h = (k/16)*8 + k%8.
    unsigned short* bdst = &bfrag[0][dn >> 4][(dn & 15) + 16 * (dj & 1)][(dj >> 1) * 8];
    const int bufStride = 4 * 32 * 16;    // halves per buffer

    // A-fragment addresses (doc'd 16-bit A layout: two 8-half runs per lane).
    const unsigned short* aptr[2];
    #pragma unroll
    for (int mt = 0; mt < 2; ++mt)
        aptr[mt] = xb + (size_t)(w * 32 + mt * 16 + (lane & 15)) * 8192
                      + ((lane >> 4) ? 8 : 0);

    int buf = 0;
    for (int k0 = kbeg; k0 < kend; k0 += 32) {
        // ---- dequant 8 indices -> one 16B fragment store -------------------
        const int* ip = irow + k0;
        int4 i0 = *(const int4*)ip;
        int4 i1 = *(const int4*)(ip + 4);
        __builtin_prefetch(ip + 32, 0, 1);              // global_prefetch_b8
        unsigned int d0 = (unsigned int)cb[i0.x & 255] | ((unsigned int)cb[i0.y & 255] << 16);
        unsigned int d1 = (unsigned int)cb[i0.z & 255] | ((unsigned int)cb[i0.w & 255] << 16);
        unsigned int d2 = (unsigned int)cb[i1.x & 255] | ((unsigned int)cb[i1.y & 255] << 16);
        unsigned int d3 = (unsigned int)cb[i1.z & 255] | ((unsigned int)cb[i1.w & 255] << 16);
        *(uint4*)(bdst + buf * bufStride) = make_uint4(d0, d1, d2, d3);
        __syncthreads();

        // ---- A fragments straight from bf16 x (L2-resident) ----------------
        v16bf a[2];
        #pragma unroll
        for (int mt = 0; mt < 2; ++mt) {
            Bits256 tmp;
            tmp.lo = *(const uint4*)(aptr[mt] + k0);
            tmp.hi = *(const uint4*)(aptr[mt] + k0 + 16);
            a[mt] = __builtin_bit_cast(v16bf, tmp);
        }

        // ---- 8x v_wmma_f32_16x16x32_bf16 ----------------------------------
        #pragma unroll
        for (int nt = 0; nt < 4; ++nt) {
            v16bf b = *(const v16bf*)&bfrag[buf][nt][lane][0];
            acc[0][nt] = __builtin_amdgcn_wmma_f32_16x16x32_bf16(
                false, a[0], false, b, (short)0, acc[0][nt], false, false);
            acc[1][nt] = __builtin_amdgcn_wmma_f32_16x16x32_bf16(
                false, a[1], false, b, (short)0, acc[1][nt], false, false);
        }
        buf ^= 1;
    }

    // ---- epilogue: C layout m = r + 8*(lane/16), n = lane%16 ---------------
    const int nl = lane & 15;
    const int mh = (lane >> 4) * 8;
    #pragma unroll
    for (int mt = 0; mt < 2; ++mt)
        #pragma unroll
        for (int nt = 0; nt < 4; ++nt) {
            float* dst = y + (size_t)(w * 32 + mt * 16 + mh) * 8192
                           + (n0 + nt * 16 + nl);
            #pragma unroll
            for (int r = 0; r < 8; ++r)
                atomicAdd(dst + (size_t)r * 8192, acc[mt][nt][r]);
        }
}

// ---------------- sidecar: y[m,row] += x[m,col]*delta ----------------------
__global__ void __launch_bounds__(256)
k_side(const float* __restrict__ x, const int* __restrict__ rows,
       const int* __restrict__ cols, const float* __restrict__ deltas,
       float* __restrict__ y) {
    const int e = blockIdx.x;
    const int m = threadIdx.x;
    const int r = rows[e], c = cols[e];
    const float d = deltas[e];
    atomicAdd(y + (size_t)m * 8192 + r, x[(size_t)m * 8192 + c] * d);
}

extern "C" void kernel_launch(void* const* d_in, const int* in_sizes, int n_in,
                              void* d_out, int out_size, void* d_ws, size_t ws_size,
                              hipStream_t stream) {
    const float* x        = (const float*)d_in[0];
    const float* codebook = (const float*)d_in[1];
    const int*   indices  = (const int*)d_in[2];
    const int*   srows    = (const int*)d_in[3];
    const int*   scols    = (const int*)d_in[4];
    const float* sdeltas  = (const float*)d_in[5];
    const float* U        = (const float*)d_in[6];
    const float* s        = (const float*)d_in[7];
    const float* Vt       = (const float*)d_in[8];
    const float* bias     = (const float*)d_in[9];
    float* y = (float*)d_out;

    unsigned short* xb = (unsigned short*)d_ws;                    // 4 MB bf16 x
    float* T = (float*)((char*)d_ws + (size_t)256 * 8192 * 2);     // 32 KB

    k_cvt <<<2048, 256, 0, stream>>>(x, xb);
    k_T   <<<256, 256, 0, stream>>>(x, Vt, s, T);
    k_init<<<dim3(32, 256), 256, 0, stream>>>(T, U, bias, y);
    k_gemm<<<dim3(128, 2), 256, 0, stream>>>(xb, indices, codebook, y);
    k_side<<<8192, 256, 0, stream>>>(x, srows, scols, sdeltas, y);
}
